// GCN_33758442947404
// MI455X (gfx1250) — matrix operand order repaired and checked by reading, not compiled, not run on hardware
//
#include <hip/hip_runtime.h>
#include <hip/hip_bf16.h>

#define N_NODES 65536
#define N_EDGES 1048576
#define D_IN    1024
#define D_HID   64
#define N_CLS   32

typedef __attribute__((ext_vector_type(16))) __bf16 v16bf;
typedef __attribute__((ext_vector_type(8)))  float  v8f;

union BF16Frag {
    v16bf        v;
    unsigned int w[8];
};

__device__ __forceinline__ unsigned short f2bf(float f) {
    // round-to-nearest-even fp32 -> bf16
    unsigned int u = __float_as_uint(f);
    u += 0x7fffu + ((u >> 16) & 1u);
    return (unsigned short)(u >> 16);
}

// ---------------------------------------------------------------- utilities
__global__ void fill_f32(float* __restrict__ p, float v, int n) {
    int i = blockIdx.x * blockDim.x + threadIdx.x;
    if (i < n) p[i] = v;
}

__global__ void deg_accum(const long long* __restrict__ ei, float* __restrict__ deg) {
    int e = blockIdx.x * blockDim.x + threadIdx.x;
    if (e < N_EDGES) {
        int d = (int)ei[N_EDGES + e];
        atomicAdd(&deg[d], 1.0f);
    }
}

__global__ void deg_finish(const float* __restrict__ deg,
                           float* __restrict__ dinv, float* __restrict__ invdeg) {
    int i = blockIdx.x * blockDim.x + threadIdx.x;
    if (i < N_NODES) {
        float dg = deg[i];
        dinv[i]   = rsqrtf(dg);
        invdeg[i] = 1.0f / dg;
    }
}

// ---------------------------------------------------------------- GEMM1: [65536,1024]x[1024,64] bf16 WMMA
// Block = 256 threads (8 waves). Tile: 128 rows x 64 cols. K-step 32.
__global__ __launch_bounds__(256) void gemm1_wmma(const float* __restrict__ x,
                                                  const float* __restrict__ W1,
                                                  float* __restrict__ xw1) {
    __shared__ unsigned short At[128 * 36];  // A tile bf16 [row][k], stride 36
    __shared__ unsigned short Bt[64 * 36];   // W1 tile bf16 transposed [n][k], stride 36

    const int tid     = threadIdx.x;
    const int wave    = tid >> 5;
    const int lane    = tid & 31;
    const int half    = lane >> 4;
    const int l15     = lane & 15;
    const int rowBase = blockIdx.x * 128;

    v8f acc[4] = {};

    for (int k0 = 0; k0 < D_IN; k0 += 32) {
        // stage A: 128x32 fp32 -> bf16 LDS (coalesced float4 loads)
        #pragma unroll
        for (int j = 0; j < 4; ++j) {
            int idx = j * 256 + tid;
            int f   = idx * 4;              // flat element in 128x32 tile
            int r   = f >> 5;
            int kk  = f & 31;               // multiple of 4
            const float4 v4 = *reinterpret_cast<const float4*>(
                &x[(size_t)(rowBase + r) * D_IN + k0 + kk]);
            unsigned int p0 = (unsigned int)f2bf(v4.x) | ((unsigned int)f2bf(v4.y) << 16);
            unsigned int p1 = (unsigned int)f2bf(v4.z) | ((unsigned int)f2bf(v4.w) << 16);
            *reinterpret_cast<unsigned int*>(&At[r * 36 + kk])     = p0;
            *reinterpret_cast<unsigned int*>(&At[r * 36 + kk + 2]) = p1;
        }
        // stage B: W1[k0..k0+31][0..63] -> transposed bf16 LDS [n][k]
        #pragma unroll
        for (int j = 0; j < 2; ++j) {
            int idx = j * 256 + tid;
            int f   = idx * 4;
            int kr  = f >> 6;
            int n   = f & 63;
            const float4 v4 = *reinterpret_cast<const float4*>(&W1[(k0 + kr) * D_HID + n]);
            Bt[(n + 0) * 36 + kr] = f2bf(v4.x);
            Bt[(n + 1) * 36 + kr] = f2bf(v4.y);
            Bt[(n + 2) * 36 + kr] = f2bf(v4.z);
            Bt[(n + 3) * 36 + kr] = f2bf(v4.w);
        }
        __syncthreads();

        // A fragment: 16x32 bf16 (lane=row%16, K pairs per VGPR, +8*(lane/16))
        BF16Frag a;
        #pragma unroll
        for (int v = 0; v < 8; ++v) {
            int k = 2 * (v & 3) + 16 * (v >> 2) + 8 * half;
            a.w[v] = *reinterpret_cast<const unsigned int*>(&At[(wave * 16 + l15) * 36 + k]);
        }
        #pragma unroll
        for (int c = 0; c < 4; ++c) {
            // B fragment: 32x16 bf16 (lane=col%16, K pair 2*(v&3) + 16*(v>>2) + 8*(lane/16))
            BF16Frag b;
            #pragma unroll
            for (int v = 0; v < 8; ++v) {
                int k = 2 * (v & 3) + 16 * (v >> 2) + 8 * half;
                b.w[v] = *reinterpret_cast<const unsigned int*>(&Bt[(c * 16 + l15) * 36 + k]);
            }
            acc[c] = __builtin_amdgcn_wmma_f32_16x16x32_bf16(
                false, a.v, false, b.v, (short)0, acc[c], false, false);
        }
        __syncthreads();
    }

    // store D: VGPR r -> row (lane/16)*8 + r, col = lane%16
    #pragma unroll
    for (int c = 0; c < 4; ++c) {
        #pragma unroll
        for (int r = 0; r < 8; ++r) {
            int row = rowBase + wave * 16 + half * 8 + r;
            int col = c * 16 + l15;
            xw1[(size_t)row * D_HID + col] = acc[c][r];
        }
    }
}

// ---------------------------------------------------------------- edge scatter: agg[dst] += feat[src]*norm
template <int D>
__global__ __launch_bounds__(256) void agg_edges(const long long* __restrict__ ei,
                                                 const float* __restrict__ feat,
                                                 const float* __restrict__ dinv,
                                                 float* __restrict__ agg) {
    int e    = blockIdx.x * 8 + (threadIdx.x >> 5);  // one wave32 per edge
    int lane = threadIdx.x & 31;
    if (e >= N_EDGES) return;
    int   s   = (int)ei[e];
    int   d   = (int)ei[N_EDGES + e];
    float nrm = dinv[s] * dinv[d];
    #pragma unroll
    for (int i = 0; i < D / 32; ++i) {
        int f = lane + 32 * i;
        atomicAdd(&agg[(size_t)d * D + f], feat[(size_t)s * D + f] * nrm);
    }
}

// ---------------------------------------------------------------- layer-1 epilogue: relu(agg + xw/deg + b1) -> bf16
__global__ void post1(const float* __restrict__ agg1, const float* __restrict__ xw1,
                      const float* __restrict__ invdeg, const float* __restrict__ b1,
                      unsigned short* __restrict__ hbf) {
    int idx = blockIdx.x * blockDim.x + threadIdx.x;  // N_NODES*64 threads
    int row = idx >> 6;
    int col = idx & 63;
    float v = agg1[idx] + xw1[idx] * invdeg[row] + b1[col];
    hbf[idx] = f2bf(fmaxf(v, 0.0f));
}

// ---------------------------------------------------------------- GEMM2: [65536,64]bf16 x [64,32] bf16 WMMA
__global__ __launch_bounds__(256) void gemm2_wmma(const unsigned short* __restrict__ hbf,
                                                  const float* __restrict__ W2,
                                                  float* __restrict__ xw2) {
    __shared__ unsigned short Bt[32 * 68];  // W2 transposed bf16 [n][k], stride 68

    const int tid     = threadIdx.x;
    const int wave    = tid >> 5;
    const int lane    = tid & 31;
    const int half    = lane >> 4;
    const int l15     = lane & 15;
    const int rowBase = blockIdx.x * 128;

    #pragma unroll
    for (int j = 0; j < 8; ++j) {
        int f  = j * 256 + tid;   // f = k*32 + n over W2 [64][32]
        int kr = f >> 5;
        int n  = f & 31;
        Bt[n * 68 + kr] = f2bf(W2[f]);
    }
    __syncthreads();

    v8f acc[2] = {};
    #pragma unroll
    for (int ks = 0; ks < D_HID; ks += 32) {
        BF16Frag a;
        int row = rowBase + wave * 16 + l15;
        #pragma unroll
        for (int v = 0; v < 8; ++v) {
            int k = ks + 2 * (v & 3) + 16 * (v >> 2) + 8 * half;
            a.w[v] = *reinterpret_cast<const unsigned int*>(&hbf[(size_t)row * D_HID + k]);
        }
        #pragma unroll
        for (int c = 0; c < 2; ++c) {
            BF16Frag b;
            #pragma unroll
            for (int v = 0; v < 8; ++v) {
                int k = ks + 2 * (v & 3) + 16 * (v >> 2) + 8 * half;
                b.w[v] = *reinterpret_cast<const unsigned int*>(&Bt[(c * 16 + l15) * 68 + k]);
            }
            acc[c] = __builtin_amdgcn_wmma_f32_16x16x32_bf16(
                false, a.v, false, b.v, (short)0, acc[c], false, false);
        }
    }
    #pragma unroll
    for (int c = 0; c < 2; ++c) {
        #pragma unroll
        for (int r = 0; r < 8; ++r) {
            int row = rowBase + wave * 16 + half * 8 + r;
            int col = c * 16 + l15;
            xw2[(size_t)row * N_CLS + col] = acc[c][r];
        }
    }
}

// ---------------------------------------------------------------- layer-2 epilogue + log_softmax (wave32 per row)
__global__ __launch_bounds__(256) void post2_logsoftmax(const float* __restrict__ agg2,
                                                        const float* __restrict__ xw2,
                                                        const float* __restrict__ invdeg,
                                                        const float* __restrict__ b2,
                                                        float* __restrict__ out) {
    int row  = blockIdx.x * 8 + (threadIdx.x >> 5);
    int lane = threadIdx.x & 31;
    int idx  = row * N_CLS + lane;
    float z  = agg2[idx] + xw2[idx] * invdeg[row] + b2[lane];
    float m = z;
    #pragma unroll
    for (int off = 16; off > 0; off >>= 1) m = fmaxf(m, __shfl_xor(m, off, 32));
    float ex = __expf(z - m);
    float s  = ex;
    #pragma unroll
    for (int off = 16; off > 0; off >>= 1) s += __shfl_xor(s, off, 32);
    out[idx] = z - m - __logf(s);
}

// ---------------------------------------------------------------- launcher
extern "C" void kernel_launch(void* const* d_in, const int* in_sizes, int n_in,
                              void* d_out, int out_size, void* d_ws, size_t ws_size,
                              hipStream_t stream) {
    (void)in_sizes; (void)n_in; (void)out_size; (void)ws_size;
    const float*     x   = (const float*)d_in[0];
    const long long* ei  = (const long long*)d_in[1];  // [2, E] int64
    const float*     W1  = (const float*)d_in[2];
    const float*     b1  = (const float*)d_in[3];
    const float*     W2  = (const float*)d_in[4];
    const float*     b2  = (const float*)d_in[5];
    float*           out = (float*)d_out;

    char* ws  = (char*)d_ws;
    size_t off = 0;
    auto carve = [&](size_t bytes) {
        char* p = ws + off;
        off += (bytes + 255) & ~(size_t)255;
        return p;
    };
    float*          xw1    = (float*)carve((size_t)N_NODES * D_HID * 4);
    float*          agg1   = (float*)carve((size_t)N_NODES * D_HID * 4);
    unsigned short* hbf    = (unsigned short*)carve((size_t)N_NODES * D_HID * 2);
    float*          xw2    = (float*)carve((size_t)N_NODES * N_CLS * 4);
    float*          agg2   = (float*)carve((size_t)N_NODES * N_CLS * 4);
    float*          deg    = (float*)carve((size_t)N_NODES * 4);
    float*          dinv   = (float*)carve((size_t)N_NODES * 4);
    float*          invdeg = (float*)carve((size_t)N_NODES * 4);

    // init (self-loop degree 1.0; zero accumulators every call)
    fill_f32<<<N_NODES / 256, 256, 0, stream>>>(deg, 1.0f, N_NODES);
    fill_f32<<<(N_NODES * D_HID) / 256, 256, 0, stream>>>(agg1, 0.0f, N_NODES * D_HID);
    fill_f32<<<(N_NODES * N_CLS) / 256, 256, 0, stream>>>(agg2, 0.0f, N_NODES * N_CLS);

    deg_accum<<<N_EDGES / 256, 256, 0, stream>>>(ei, deg);
    deg_finish<<<N_NODES / 256, 256, 0, stream>>>(deg, dinv, invdeg);

    gemm1_wmma<<<N_NODES / 128, 256, 0, stream>>>(x, W1, xw1);
    agg_edges<D_HID><<<N_EDGES / 8, 256, 0, stream>>>(ei, xw1, dinv, agg1);
    post1<<<(N_NODES * D_HID) / 256, 256, 0, stream>>>(agg1, xw1, invdeg, b1, hbf);

    gemm2_wmma<<<N_NODES / 128, 256, 0, stream>>>(hbf, W2, xw2);
    agg_edges<N_CLS><<<N_EDGES / 8, 256, 0, stream>>>(ei, xw2, dinv, agg2);
    post2_logsoftmax<<<N_NODES / 8, 256, 0, stream>>>(agg2, xw2, invdeg, b2, out);
}
